// Drnet_2216203124968
// MI455X (gfx1250) — matrix operand order repaired and checked by reading, not compiled, not run on hardware
//
#include <hip/hip_runtime.h>
#include <hip/hip_bf16.h>
#include <cstdint>

// ---------------------------------------------------------------------------
// DRNet mixture-of-GRU-experts, CDNA5 (gfx1250) wave32 + WMMA f16 implementation.
//   B=65536, H=50, T=7, E=5.
// Pipeline:
//   k_count / k_scan / k_scatter : bucket-sort batch indices by treatment t
//   k_prep                       : f32 weights -> f16 WMMA-B-layout tiles in d_ws
//   k_main                       : 16-row tiles, all matmuls via v_wmma_f32_16x16x32_f16
// ---------------------------------------------------------------------------

typedef __attribute__((ext_vector_type(16))) _Float16 v16h;
typedef __attribute__((ext_vector_type(8)))  float    v8f;

#define NBATCH 65536
#define HH 50
#define TT 7
#define NE 5
#define XC 51        // 1 + H columns of x
#define KP 64        // padded K (50 -> 64) = 2 k-blocks of 32
#define KBLK 2
#define NT3 10       // n-tiles covering 150 outputs (padded 160)
#define NTH 4        // n-tiles covering 50 outputs  (padded 64)
#define G3 160       // gi/gh row stride (f16)

// ---- workspace layout (bytes) ----
// [0,32)   counts[8]   [32,64) cursors[8]
// [64, 64+4*NBATCH)    perm
// [WS_WT_OFF, +400KB)  f16 weight tiles
#define WS_PERM_OFF 64
#define WS_WT_OFF   262208          // 64 + 4*65536, 32B aligned
// f16 tile-region offsets (in f16 elements) and per-expert strides
#define WT_IH  0
#define WT_HH  51200
#define WT_FC  102400
#define WT_S1A 122880
#define WT_S1B 143360
#define WT_L1A 163840
#define WT_L1B 184320
#define ES_IH  10240                // 10 ntiles * 2 kblocks * 512
#define ES_H4  4096                 // 4 ntiles * 2 kblocks * 512

__device__ __forceinline__ int bucket_of(float t, float p1, float p2, float p3, float p4) {
  return (t >= p1) + (t >= p2) + (t >= p3) + (t >= p4);   // searchsorted(right), max 4
}

// Wave-local LDS fence: same-wave DS ops complete in order; the memory clobber
// pins program order, s_wait_dscnt drains outstanding LDS before reuse.
__device__ __forceinline__ void wsync() {
  asm volatile("s_wait_dscnt 0" ::: "memory");
}

// A-operand (16x32 f16) from an LDS row-major [16][KP] f16 tile.
// ISA 7.12.2: lane group g=lane>>4: elements j<8 -> K=k0+8g+j ; j>=8 -> K=k0+16+8g+(j-8)
__device__ __forceinline__ v16h load_A16(const _Float16* base, int lane, int k0) {
  const int m = lane & 15, g = lane >> 4;
  const _Float16* p = base + m * KP + k0 + g * 8;
  v16h a;
#pragma unroll
  for (int j = 0; j < 8; ++j) { a[j] = p[j]; a[j + 8] = p[16 + j]; }
  return a;
}

// B-operand (32x16 f16): pre-swizzled tile, lane-major [32][16] -> 2x b128 per lane.
__device__ __forceinline__ v16h load_B16(const _Float16* tile, int lane) {
  const _Float16* p = tile + lane * 16;
  v16h b;
#pragma unroll
  for (int j = 0; j < 16; ++j) b[j] = p[j];
  return b;
}

__device__ __forceinline__ v8f wmma16(v16h a, v16h b, v8f c) {
  return __builtin_amdgcn_wmma_f32_16x16x32_f16(false, a, false, b, (short)0, c, false, false);
}

// one 16x16 output tile over the full (padded) K=64 reduction
__device__ __forceinline__ v8f mm_tile(const _Float16* Alds, const _Float16* Bt, int lane) {
  v8f c = {0.f, 0.f, 0.f, 0.f, 0.f, 0.f, 0.f, 0.f};
#pragma unroll
  for (int tk = 0; tk < KBLK; ++tk)
    c = wmma16(load_A16(Alds, lane, tk * 32), load_B16(Bt + tk * 512, lane), c);
  return c;
}

// ---------------------------------------------------------------------------
__global__ void k_count(const float* __restrict__ x, const float* __restrict__ pt,
                        int* __restrict__ counts) {
  int b = blockIdx.x * blockDim.x + threadIdx.x;
  if (b >= NBATCH) return;
  float t = x[(size_t)b * XC];
  atomicAdd(&counts[bucket_of(t, pt[1], pt[2], pt[3], pt[4])], 1);
}

__global__ void k_scan(const int* __restrict__ counts, int* __restrict__ cursors) {
  if (threadIdx.x == 0) {
    int c = 0;
    for (int e = 0; e < NE; ++e) { cursors[e] = c; c += counts[e]; }
  }
}

__global__ void k_scatter(const float* __restrict__ x, const float* __restrict__ pt,
                          int* cursors, int* __restrict__ perm) {
  int b = blockIdx.x * blockDim.x + threadIdx.x;
  if (b >= NBATCH) return;
  float t = x[(size_t)b * XC];
  int bk = bucket_of(t, pt[1], pt[2], pt[3], pt[4]);
  int pos = atomicAdd(&cursors[bk], 1);
  perm[pos] = b;
}

// f32 weights -> f16 WMMA-B tiles. One block (32 lanes) per (expert, matrix, tile).
__global__ void k_prep(const float* __restrict__ Wih, const float* __restrict__ Whh,
                       const float* __restrict__ Wfc, const float* __restrict__ Ws1,
                       const float* __restrict__ Wl1, _Float16* __restrict__ wt) {
  const int cnt[7]  = {20, 20, 8, 8, 8, 8, 8};
  const int Ns[7]   = {150, 150, 50, 50, 50, 50, 50};
  const int koff[7] = {0, 0, 0, 0, 50, 0, 50};
  const int kstr[7] = {50, 50, 50, 100, 100, 100, 100};
  const int dst0[7] = {WT_IH, WT_HH, WT_FC, WT_S1A, WT_S1B, WT_L1A, WT_L1B};
  const int estr[7] = {ES_IH, ES_IH, ES_H4, ES_H4, ES_H4, ES_H4, ES_H4};
  int bid = blockIdx.x;
  int e = bid / 80, tl = bid % 80;
  int m = 0, acc = 0;
  while (tl >= acc + cnt[m]) { acc += cnt[m]; ++m; }
  int tt = tl - acc;
  int tn = tt / KBLK, tk = tt % KBLK;
  const float* src = (m == 0) ? Wih + e * 7500
                   : (m == 1) ? Whh + e * 7500
                   : (m == 2) ? Wfc + e * 2500
                   : (m <= 4) ? Ws1 + e * 5000
                              : Wl1 + e * 5000;
  _Float16* dst = wt + dst0[m] + e * estr[m] + (tn * KBLK + tk) * 512;
  int l = threadIdx.x;
  int n  = tn * 16 + (l & 15);
  int kb = tk * 32 + (l >> 4) * 8;       // B layout: lane group g: j<8 -> K=8g+j ; j>=8 -> K=16+8g+(j-8)
#pragma unroll
  for (int j = 0; j < 16; ++j) {
    int k = kb + (j < 8 ? j : j + 8);
    float v = (n < Ns[m] && k < HH) ? src[n * kstr[m] + koff[m] + k] : 0.f;
    dst[l * 16 + j] = (_Float16)v;
  }
}

// ---------------------------------------------------------------------------
struct WaveLDS {                 // 31616 bytes / wave
  _Float16 xf[16 * KP];          // gathered features (f16, K-padded)
  _Float16 h[16 * KP];           // GRU hidden state
  _Float16 tre[16 * KP];         // treatment embedding t*W_t + b_t
  _Float16 ctx[16 * KP];         // attention context (post-softmax)
  _Float16 gi[16 * G3];          // input-proj gates (3H, padded 160)
  _Float16 gh[16 * G3];          // hidden-proj gates
  float tr[16 * HH];             // tanh attention features (one step)
  float s1[16 * HH];             // head hidden
  float sb[16 * HH];             // time-invariant st base
  float ca[16 * HH];             // online-softmax ctx accumulator
  float rowt[16], alp[16], bet[16], den[16];
  int rowb[16], rowp[16];
};

__global__ void __launch_bounds__(64) k_main(
    const float* __restrict__ x,   const float* __restrict__ b_ih,
    const float* __restrict__ b_hh, const float* __restrict__ b_fc,
    const float* __restrict__ u,    const float* __restrict__ W_t,
    const float* __restrict__ b_t,  const float* __restrict__ bs1,
    const float* __restrict__ Ws2,  const float* __restrict__ bs2,
    const float* __restrict__ bl1,  const float* __restrict__ Wl2,
    const float* __restrict__ bl2,  const float* __restrict__ pt,
    const int* __restrict__ perm,   const _Float16* __restrict__ wt,
    float* __restrict__ out) {
  __shared__ WaveLDS wls[2];
  const int lane = threadIdx.x & 31;
  const int wave = threadIdx.x >> 5;
  WaveLDS& W = wls[wave];
  const int tile = blockIdx.x * 2 + wave;
  const int b0 = tile * 16;

  float* rep  = out;
  float* outs = out + (size_t)NBATCH * TT * HH;
  float* outy = outs + (size_t)NBATCH * TT;

  const float p1 = pt[1], p2 = pt[2], p3 = pt[3], p4 = pt[4];

  if (lane < 16) {
    int src = perm[b0 + lane];
    W.rowp[lane] = src;
    float tv = x[(size_t)src * XC];
    W.rowt[lane] = tv;
    W.rowb[lane] = bucket_of(tv, p1, p2, p3, p4);
  }
  wsync();
  for (int i = lane; i < 16 * KP; i += 32) {               // gather + f16 convert
    int r = i >> 6, c = i & 63;
    W.xf[i] = (c < HH) ? (_Float16)x[(size_t)W.rowp[r] * XC + 1 + c] : (_Float16)0.f;
  }
  wsync();

  const int g = lane >> 4, nl = lane & 15;

  for (int e = 0; e < NE; ++e) {                           // bucket-sorted: usually 1 pass
    bool mine = (lane < 16) && (W.rowb[lane] == e);
    if (!__any(mine)) continue;                            // wave-uniform skip (EXEC stays full)

    const _Float16* Bih = wt + WT_IH  + e * ES_IH;
    const _Float16* Bhh = wt + WT_HH  + e * ES_IH;
    const _Float16* Bfc = wt + WT_FC  + e * ES_H4;
    const _Float16* Ba  = wt + WT_S1A + e * ES_H4;
    const _Float16* Bb  = wt + WT_S1B + e * ES_H4;
    const _Float16* Bla = wt + WT_L1A + e * ES_H4;
    const _Float16* Blb = wt + WT_L1B + e * ES_H4;
    const float* bih  = b_ih + e * 150;
    const float* bhh  = b_hh + e * 150;
    const float* bfc  = b_fc + e * HH;
    const float* ue   = u    + e * HH;
    const float* wte  = W_t  + e * HH;
    const float* bte  = b_t  + e * HH;
    const float* bs1e = bs1  + e * HH;
    const float* ws2e = Ws2  + e * HH;
    const float  bs2e = bs2[e];
    const float* bl1e = bl1  + e * HH;
    const float* wl2e = Wl2  + e * HH;
    const float  bl2e = bl2[e];

    for (int i = lane; i < 16 * KP; i += 32) {             // h=0, treat embedding
      int r = i >> 6, c = i & 63;
      W.h[i]   = (_Float16)0.f;
      W.tre[i] = (c < HH) ? (_Float16)(W.rowt[r] * wte[c] + bte[c]) : (_Float16)0.f;
    }
    for (int i = lane; i < 16 * HH; i += 32) W.ca[i] = 0.f;
    wsync();

    // gi = x_feat @ W_ih^T + b_ih
#pragma unroll
    for (int tn = 0; tn < NT3; ++tn) {
      v8f c = mm_tile(W.xf, Bih + tn * (KBLK * 512), lane);
      int n = tn * 16 + nl;
      float bias = (n < 150) ? bih[n] : 0.f;
#pragma unroll
      for (int v = 0; v < 8; ++v) W.gi[(v + 8 * g) * G3 + n] = (_Float16)(c[v] + bias);
    }
    // st base = treat @ Ws1[:,50:]^T + bs1   (loop-invariant, hoisted)
#pragma unroll
    for (int tn = 0; tn < NTH; ++tn) {
      v8f c = mm_tile(W.tre, Bb + tn * (KBLK * 512), lane);
      int n = tn * 16 + nl;
      if (n < HH) {
#pragma unroll
        for (int v = 0; v < 8; ++v) W.sb[(v + 8 * g) * HH + n] = c[v] + bs1e[n];
      }
    }
    wsync();

    float mrun = -3.0e38f, drun = 0.f;                     // online softmax state (rows 0..15)

    for (int t = 0; t < TT; ++t) {
      // gh = h @ W_hh^T + b_hh
#pragma unroll
      for (int tn = 0; tn < NT3; ++tn) {
        v8f c = mm_tile(W.h, Bhh + tn * (KBLK * 512), lane);
        int n = tn * 16 + nl;
        float bias = (n < 150) ? bhh[n] : 0.f;
#pragma unroll
        for (int v = 0; v < 8; ++v) W.gh[(v + 8 * g) * G3 + n] = (_Float16)(c[v] + bias);
      }
      wsync();
      // GRU pointwise update + rep_s store
      for (int i = lane; i < 16 * HH; i += 32) {
        int r = i / HH, c = i - r * HH;
        float ir = (float)W.gi[r * G3 + c], iz = (float)W.gi[r * G3 + 50 + c],
              in = (float)W.gi[r * G3 + 100 + c];
        float hr = (float)W.gh[r * G3 + c], hz = (float)W.gh[r * G3 + 50 + c],
              hn = (float)W.gh[r * G3 + 100 + c];
        float rr = 1.f / (1.f + __expf(-(ir + hr)));
        float zz = 1.f / (1.f + __expf(-(iz + hz)));
        float nn = tanhf(in + rr * hn);
        float hold = (float)W.h[r * KP + c];
        float hnew = (1.f - zz) * nn + zz * hold;
        W.h[r * KP + c] = (_Float16)hnew;
        if (W.rowb[r] == e) rep[(size_t)W.rowp[r] * (TT * HH) + t * HH + c] = hnew;
      }
      wsync();
      // tr = tanh(h @ W_fc^T + b_fc)
#pragma unroll
      for (int tn = 0; tn < NTH; ++tn) {
        v8f c = mm_tile(W.h, Bfc + tn * (KBLK * 512), lane);
        int n = tn * 16 + nl;
        if (n < HH) {
#pragma unroll
          for (int v = 0; v < 8; ++v) W.tr[(v + 8 * g) * HH + n] = tanhf(c[v] + bfc[n]);
        }
      }
      wsync();
      // attention score + online softmax rescale factors
      if (lane < 16) {
        float s = 0.f;
        for (int k = 0; k < HH; ++k) s += W.tr[lane * HH + k] * ue[k];
        float mn = fmaxf(mrun, s);
        float a = __expf(mrun - mn), bt2 = __expf(s - mn);
        W.alp[lane] = a; W.bet[lane] = bt2;
        drun = drun * a + bt2;
        mrun = mn;
      }
      wsync();
      for (int i = lane; i < 16 * HH; i += 32) {           // ctx accumulator update
        int r = i / HH;
        W.ca[i] = W.ca[i] * W.alp[r] + W.bet[r] * W.tr[i];
      }
      // short-term head: s1 = relu(stbase + h @ Ws1[:,:50]^T)
#pragma unroll
      for (int tn = 0; tn < NTH; ++tn) {
        v8f c = mm_tile(W.h, Ba + tn * (KBLK * 512), lane);
        int n = tn * 16 + nl;
        if (n < HH) {
#pragma unroll
          for (int v = 0; v < 8; ++v) {
            int m = v + 8 * g;
            W.s1[m * HH + n] = fmaxf(0.f, c[v] + W.sb[m * HH + n]);
          }
        }
      }
      wsync();
      if (lane < 16) {
        float s = bs2e;
        for (int k = 0; k < HH; ++k) s += W.s1[lane * HH + k] * ws2e[k];
        if (W.rowb[lane] == e) outs[(size_t)W.rowp[lane] * TT + t] = s;
      }
      wsync();
    }

    // finalize ctx = ca / denom, f16-pad for the A operand
    if (lane < 16) W.den[lane] = drun;
    wsync();
    for (int i = lane; i < 16 * KP; i += 32) {
      int r = i >> 6, c = i & 63;
      W.ctx[i] = (c < HH) ? (_Float16)(W.ca[r * HH + c] / W.den[r]) : (_Float16)0.f;
    }
    wsync();
    // long-term head: l1 = relu(ctx@Wl1[:,:50]^T + treat@Wl1[:,50:]^T + bl1)
#pragma unroll
    for (int tn = 0; tn < NTH; ++tn) {
      v8f c = {0.f, 0.f, 0.f, 0.f, 0.f, 0.f, 0.f, 0.f};
#pragma unroll
      for (int tk = 0; tk < KBLK; ++tk)
        c = wmma16(load_A16(W.ctx, lane, tk * 32),
                   load_B16(Bla + tn * (KBLK * 512) + tk * 512, lane), c);
#pragma unroll
      for (int tk = 0; tk < KBLK; ++tk)
        c = wmma16(load_A16(W.tre, lane, tk * 32),
                   load_B16(Blb + tn * (KBLK * 512) + tk * 512, lane), c);
      int n = tn * 16 + nl;
      if (n < HH) {
#pragma unroll
        for (int v = 0; v < 8; ++v) W.s1[(v + 8 * g) * HH + n] = fmaxf(0.f, c[v] + bl1e[n]);
      }
    }
    wsync();
    if (lane < 16) {
      float s = bl2e;
      for (int k = 0; k < HH; ++k) s += W.s1[lane * HH + k] * wl2e[k];
      if (W.rowb[lane] == e) outy[W.rowp[lane]] = s;
    }
    wsync();
  }
}

// ---------------------------------------------------------------------------
extern "C" void kernel_launch(void* const* d_in, const int* in_sizes, int n_in,
                              void* d_out, int out_size, void* d_ws, size_t ws_size,
                              hipStream_t stream) {
  (void)in_sizes; (void)n_in; (void)out_size; (void)ws_size;
  const float* x   = (const float*)d_in[0];
  const float* Wih = (const float*)d_in[1];
  const float* Whh = (const float*)d_in[2];
  const float* bih = (const float*)d_in[3];
  const float* bhh = (const float*)d_in[4];
  const float* Wfc = (const float*)d_in[5];
  const float* bfc = (const float*)d_in[6];
  const float* u   = (const float*)d_in[7];
  const float* Wt  = (const float*)d_in[8];
  const float* bt  = (const float*)d_in[9];
  const float* Ws1 = (const float*)d_in[10];
  const float* bs1 = (const float*)d_in[11];
  const float* Ws2 = (const float*)d_in[12];
  const float* bs2 = (const float*)d_in[13];
  const float* Wl1 = (const float*)d_in[14];
  const float* bl1 = (const float*)d_in[15];
  const float* Wl2 = (const float*)d_in[16];
  const float* bl2 = (const float*)d_in[17];
  const float* pt  = (const float*)d_in[18];

  int* counts  = (int*)d_ws;
  int* cursors = counts + 8;
  int* perm    = (int*)((char*)d_ws + WS_PERM_OFF);
  _Float16* wt = (_Float16*)((char*)d_ws + WS_WT_OFF);   // needs ~660 KB of d_ws

  hipMemsetAsync(d_ws, 0, 64, stream);
  k_count  <<<NBATCH / 256, 256, 0, stream>>>(x, pt, counts);
  k_scan   <<<1, 32, 0, stream>>>(counts, cursors);
  k_scatter<<<NBATCH / 256, 256, 0, stream>>>(x, pt, cursors, perm);
  k_prep   <<<NE * 80, 32, 0, stream>>>(Wih, Whh, Wfc, Ws1, Wl1, wt);
  k_main   <<<(NBATCH / 16) / 2, 64, 0, stream>>>(x, bih, bhh, bfc, u, Wt, bt, bs1, Ws2,
                                                  bs2, bl1, Wl2, bl2, pt, perm, wt,
                                                  (float*)d_out);
}
